// EncoderBlock_1726576853359
// MI455X (gfx1250) — matrix operand order repaired and checked by reading, not compiled, not run on hardware
//
#include <hip/hip_runtime.h>
#include <hip/hip_bf16.h>

// Problem constants
#define B_   16
#define S_   200
#define D_   512
#define H_   8
#define FF_  2048
#define P_   10
#define HD_  64
#define SP_  224          // S padded to a multiple of 32 for WMMA K-loops
#define EPS_ 1e-5f

// Block-GEMM tiling
#define BM_  64
#define BN_  64
#define LDH_ 40           // padded LDS row stride in halves (32 + 8 pad) = 80B
#define LDK_ 68           // padded LDS row stride in floats (64 + 4 pad) for k/v panels

typedef __attribute__((ext_vector_type(16))) _Float16 v16h;
typedef __attribute__((ext_vector_type(8)))  _Float16 v8h;
typedef __attribute__((ext_vector_type(4)))  _Float16 v4h;
typedef __attribute__((ext_vector_type(8)))  float    v8f;
typedef __attribute__((ext_vector_type(4)))  unsigned int v4u;
typedef __attribute__((ext_vector_type(8)))  int      v8i;
typedef __attribute__((ext_vector_type(4)))  int      v4i;

__device__ __forceinline__ int imin_(int a, int b) { return a < b ? a : b; }

// ---------------------------------------------------------------------------
// Tensor Data Mover 2D tile load, global -> LDS, with LDS row padding.
// D# packing per CDNA5 ISA.
// padcfg = pad_amount(code)<<25 | pad_interval(code)<<22 | pad_en<<20 | data_size<<16
//   f16 tiles (32 halves/row = 64B): interval code 3 (16 DW), amount code 3 (4 DW)
//        -> LDS stride 40 halves (80B, 16B-aligned, conflict-free)
//   f32 panels (64 floats/row = 256B): interval code 5 (64 DW), amount code 3
//        -> LDS stride 68 floats
// ---------------------------------------------------------------------------
#define TDM_F16_32_ 0x06D10000u  // 2B elems, pad 4 DW every 16 DW
#define TDM_F32_64_ 0x07520000u  // 4B elems, pad 4 DW every 64 DW

__device__ __forceinline__ void tdm_load_2d(unsigned lds_off, const void* g,
                                            int rows, int cols, int ldg, unsigned padcfg) {
  unsigned long long ga = (unsigned long long)(size_t)g;
  v4u g0;
  g0[0] = 1u;                                        // count=1, user mode
  g0[1] = lds_off;                                   // lds_addr (bytes)
  g0[2] = (unsigned)ga;                              // global_addr[31:0]
  g0[3] = (unsigned)((ga >> 32) & 0x01ffffffu) | (2u << 30);  // addr hi | type=2
  v8i g1 = {};
  g1[0] = (int)padcfg;
  g1[1] = (int)((unsigned)cols << 16);  // tensor_dim0 low16 (bits 63:48)
  g1[2] = (int)((unsigned)rows << 16);  // tensor_dim0 hi=0 | tensor_dim1 (bits 95:80)
  g1[3] = (int)((unsigned)cols << 16);  // tensor_dim1 hi=0 | tile_dim0 (bits 127:112)
  g1[4] = rows;                         // tile_dim1 (bits 143:128)
  g1[5] = ldg;                          // tensor_dim0_stride low32 (elements)
  g1[6] = 0;
  g1[7] = 0;
  v4i z4 = {};
#if defined(__clang_major__) && (__clang_major__ >= 23)
  v8i z8 = {};
  __builtin_amdgcn_tensor_load_to_lds(g0, g1, z4, z4, z8, 0);
#else
  __builtin_amdgcn_tensor_load_to_lds(g0, g1, z4, z4, 0);
#endif
}

// ---------------------------------------------------------------------------
// f16 LDS fragment gathers (native f16 tiles, no conversion in the main loop).
// A (16x32): lane row m=l&15, element e -> k = (e>>3)*16 + (l>>4)*8 + (e&7)
//            -> two 8-half (128-bit) ds loads at +0 and +16 halves
// B (32x16): lane col n=l&15, element e -> k = (e>>3)*16 + (l>>4)*8 + (e&7)
//            -> two 8-half (128-bit) ds loads at +0 and +16 halves
// ---------------------------------------------------------------------------
__device__ __forceinline__ v16h frag_a_h(const _Float16* t, int m0, int lane) {
  const _Float16* row = t + (size_t)(m0 + (lane & 15)) * LDH_ + ((lane >> 4) << 3);
  v8h lo = *(const v8h*)(row);
  v8h hi = *(const v8h*)(row + 16);
  v16h a;
#pragma unroll
  for (int e = 0; e < 8; ++e) { a[e] = lo[e]; a[8 + e] = hi[e]; }
  return a;
}

__device__ __forceinline__ v16h frag_b_h(const _Float16* t, int n0, int lane) {
  const _Float16* row = t + (size_t)(n0 + (lane & 15)) * LDH_ + ((lane >> 4) << 3);
  v8h lo = *(const v8h*)(row);
  v8h hi = *(const v8h*)(row + 16);
  v16h b;
#pragma unroll
  for (int e = 0; e < 8; ++e) { b[e] = lo[e]; b[8 + e] = hi[e]; }
  return b;
}

// ---------------------------------------------------------------------------
// f32 fragment gathers (attention panels; branchless row/k masks).
// ---------------------------------------------------------------------------
__device__ __forceinline__ v16h frag_a_f32(const float* t, size_t lda, int m0, int k0, int lane) {
  const float* row = t + (size_t)(m0 + (lane & 15)) * lda + k0 + ((lane >> 4) << 3);
  float4 q0 = *(const float4*)(row);
  float4 q1 = *(const float4*)(row + 4);
  float4 q2 = *(const float4*)(row + 16);
  float4 q3 = *(const float4*)(row + 20);
  v16h a;
  a[0]=(_Float16)q0.x;  a[1]=(_Float16)q0.y;  a[2]=(_Float16)q0.z;  a[3]=(_Float16)q0.w;
  a[4]=(_Float16)q1.x;  a[5]=(_Float16)q1.y;  a[6]=(_Float16)q1.z;  a[7]=(_Float16)q1.w;
  a[8]=(_Float16)q2.x;  a[9]=(_Float16)q2.y;  a[10]=(_Float16)q2.z; a[11]=(_Float16)q2.w;
  a[12]=(_Float16)q3.x; a[13]=(_Float16)q3.y; a[14]=(_Float16)q3.z; a[15]=(_Float16)q3.w;
  return a;
}

__device__ __forceinline__ v16h frag_a_rowmask(const float* t, size_t lda, int m0, int k0,
                                               int lane, int mmax) {
  int m = m0 + (lane & 15);
  float msk = (m < mmax) ? 1.f : 0.f;
  const float* row = t + (size_t)imin_(m, mmax - 1) * lda + k0 + ((lane >> 4) << 3);
  float4 q0 = *(const float4*)(row);
  float4 q1 = *(const float4*)(row + 4);
  float4 q2 = *(const float4*)(row + 16);
  float4 q3 = *(const float4*)(row + 20);
  v16h a;
  a[0]=(_Float16)(q0.x*msk);  a[1]=(_Float16)(q0.y*msk);  a[2]=(_Float16)(q0.z*msk);  a[3]=(_Float16)(q0.w*msk);
  a[4]=(_Float16)(q1.x*msk);  a[5]=(_Float16)(q1.y*msk);  a[6]=(_Float16)(q1.z*msk);  a[7]=(_Float16)(q1.w*msk);
  a[8]=(_Float16)(q2.x*msk);  a[9]=(_Float16)(q2.y*msk);  a[10]=(_Float16)(q2.z*msk); a[11]=(_Float16)(q2.w*msk);
  a[12]=(_Float16)(q3.x*msk); a[13]=(_Float16)(q3.y*msk); a[14]=(_Float16)(q3.z*msk); a[15]=(_Float16)(q3.w*msk);
  return a;
}

__device__ __forceinline__ v16h frag_bt_rowmask(const float* t, size_t ldw, int n0, int k0,
                                                int lane, int nmax) {
  int n = n0 + (lane & 15);
  float msk = (n < nmax) ? 1.f : 0.f;
  const float* row = t + (size_t)imin_(n, nmax - 1) * ldw + k0 + ((lane >> 4) << 3);
  float4 q0 = *(const float4*)(row);
  float4 q1 = *(const float4*)(row + 4);
  float4 q2 = *(const float4*)(row + 16);
  float4 q3 = *(const float4*)(row + 20);
  v16h b;
  b[0]=(_Float16)(q0.x*msk);  b[1]=(_Float16)(q0.y*msk);  b[2]=(_Float16)(q0.z*msk);  b[3]=(_Float16)(q0.w*msk);
  b[4]=(_Float16)(q1.x*msk);  b[5]=(_Float16)(q1.y*msk);  b[6]=(_Float16)(q1.z*msk);  b[7]=(_Float16)(q1.w*msk);
  b[8]=(_Float16)(q2.x*msk);  b[9]=(_Float16)(q2.y*msk);  b[10]=(_Float16)(q2.z*msk); b[11]=(_Float16)(q2.w*msk);
  b[12]=(_Float16)(q3.x*msk); b[13]=(_Float16)(q3.y*msk); b[14]=(_Float16)(q3.z*msk); b[15]=(_Float16)(q3.w*msk);
  return b;
}

// B from row-major Bm[K][N] (no transpose), k masked; per-element (LDS resident).
__device__ __forceinline__ v16h frag_b_kmask(const float* Bm, size_t ldb,
                                             int k0, int n0, int lane, int kmax) {
  int n  = n0 + (lane & 15);
  int kb = k0 + ((lane >> 4) << 3);
  v16h b;
#pragma unroll
  for (int e = 0; e < 16; ++e) {
    int k = kb + (e & 7) + ((e >> 3) << 4);
    float msk = (k < kmax) ? 1.f : 0.f;
    float v = Bm[(size_t)imin_(k, kmax - 1) * ldb + n];
    b[e] = (_Float16)(v * msk);
  }
  return b;
}

// ---------------------------------------------------------------------------
// f32 -> f16 pack (operand preconversion; n multiple of 4)
// ---------------------------------------------------------------------------
__global__ void cvt_f16_kernel(const float* __restrict__ X, _Float16* __restrict__ Y, int n) {
  int i = (blockIdx.x * blockDim.x + threadIdx.x) << 2;
  if (i >= n) return;
  float4 v = *(const float4*)(X + i);
  v4h o;
  o[0] = (_Float16)v.x; o[1] = (_Float16)v.y; o[2] = (_Float16)v.z; o[3] = (_Float16)v.w;
  *(v4h*)(Y + i) = o;
}

// ---------------------------------------------------------------------------
// Double-buffered TDM-staged f16 GEMM: C[M,N] = A[M,K] @ W[N,K]^T + bias (+R)(+ReLU)
// A, W are f16; tiles DMA'd as f16 (half the HBM/LDS bytes, no converts in loop).
// 8 waves/block, 64x64 block tile, wave = 16x32 (two subtiles share A frag).
// Pipeline: wait(buf) -> barrier -> prefetch(buf^1) -> compute(buf) -> barrier
// ---------------------------------------------------------------------------
template <bool RELU, bool OUT16>
__global__ __launch_bounds__(256) void wmma_gemm_nt(
    const _Float16* __restrict__ A, const _Float16* __restrict__ W,
    const float* __restrict__ bias, const float* __restrict__ R,
    void* __restrict__ Cv, int M, int N, int K) {
  __shared__ _Float16 tA[2][BM_ * LDH_];
  __shared__ _Float16 tW[2][BN_ * LDH_];

  int nb = N / BN_;
  int M0 = (int)(blockIdx.x / nb) * BM_;
  int N0 = (int)(blockIdx.x % nb) * BN_;
  int wv   = threadIdx.x >> 5;
  int lane = threadIdx.x & 31;
  int mt = wv >> 1;             // 0..3  (16-row subtile)
  int nt = (wv & 1) << 1;       // 0 or 2 (two adjacent 16-col subtiles)

  unsigned offA[2] = { (unsigned)(size_t)&tA[0][0], (unsigned)(size_t)&tA[1][0] };
  unsigned offW[2] = { (unsigned)(size_t)&tW[0][0], (unsigned)(size_t)&tW[1][0] };

  const _Float16* Ab = A + (size_t)M0 * K;
  const _Float16* Wb = W + (size_t)N0 * K;

  if (wv == 0) {                       // prologue: fill buffer 0
    tdm_load_2d(offA[0], Ab, BM_, 32, K, TDM_F16_32_);
    tdm_load_2d(offW[0], Wb, BN_, 32, K, TDM_F16_32_);
  }

  v8f acc0 = {}, acc1 = {};
  int buf = 0;
  for (int k0 = 0; k0 < K; k0 += 32, buf ^= 1) {
    if (wv == 0) __builtin_amdgcn_s_wait_tensorcnt(0);   // current buf DMA done
    __syncthreads();                                     // buf visible; prev reads done
    if (wv == 0 && (k0 + 32) < K) {                      // prefetch next while computing
      tdm_load_2d(offA[buf ^ 1], Ab + k0 + 32, BM_, 32, K, TDM_F16_32_);
      tdm_load_2d(offW[buf ^ 1], Wb + k0 + 32, BN_, 32, K, TDM_F16_32_);
    }
    v16h af = frag_a_h(tA[buf], mt * 16, lane);
    v16h b0 = frag_b_h(tW[buf], nt * 16, lane);
    v16h b1 = frag_b_h(tW[buf], (nt + 1) * 16, lane);
    acc0 = __builtin_amdgcn_wmma_f32_16x16x32_f16(false, af, false, b0, (short)0, acc0,
                                                  false, false);
    acc1 = __builtin_amdgcn_wmma_f32_16x16x32_f16(false, af, false, b1, (short)0, acc1,
                                                  false, false);
    __syncthreads();                                     // reads of buf complete
  }

  int col  = lane & 15;
  int mrow = M0 + mt * 16 + ((lane >> 4) << 3);
  int n0g  = N0 + nt * 16 + col;
  int n1g  = n0g + 16;
  float bv0 = bias[n0g];
  float bv1 = bias[n1g];
#pragma unroll
  for (int e = 0; e < 8; ++e) {
    size_t r = (size_t)(mrow + e) * N;
    float v0 = acc0[e] + bv0;
    float v1 = acc1[e] + bv1;
    if (R) { v0 += R[r + n0g]; v1 += R[r + n1g]; }
    if (RELU) { v0 = fmaxf(v0, 0.f); v1 = fmaxf(v1, 0.f); }
    if (OUT16) {
      _Float16* C = (_Float16*)Cv;
      C[r + n0g] = (_Float16)v0;
      C[r + n1g] = (_Float16)v1;
    } else {
      float* C = (float*)Cv;
      C[r + n0g] = v0;
      C[r + n1g] = v1;
    }
  }
}

// ---------------------------------------------------------------------------
// Positional bias: pb[i,j] = dot(pos[i,j,:], Wpos) + bpos
// ---------------------------------------------------------------------------
__global__ void pos_bias_kernel(const float* __restrict__ pos, const float* __restrict__ Wpos,
                                const float* __restrict__ bpos, float* __restrict__ pb) {
  int idx = blockIdx.x * blockDim.x + threadIdx.x;
  if (idx >= S_ * S_) return;
  const float* p = pos + (size_t)idx * P_;
  float s = bpos[0];
#pragma unroll
  for (int q = 0; q < P_; ++q) s += p[q] * Wpos[q];
  pb[idx] = s;
}

// ---------------------------------------------------------------------------
// scores[bh,i,j] = (k_i . k_j) * (sum(Wpos)/8) + pb[i,j]  (padded SPxSP)
// One workgroup per (b,h,half): TDM-stage the 200x64 f32 k panel into LDS,
// then 8 waves sweep 98 (7x14) WMMA tiles out of LDS.
// ---------------------------------------------------------------------------
__global__ __launch_bounds__(256) void attn_scores_kernel(
    const float* __restrict__ kv, const float* __restrict__ pb,
    const float* __restrict__ Wpos, float* __restrict__ scores) {
  __shared__ float tk[S_ * LDK_];          // 200 x 68 f32 (54.4 KB)
  int blk  = blockIdx.x;                   // 0..255
  int bh   = blk >> 1;
  int half = blk & 1;
  int b = bh / H_, h = bh % H_;
  int wv = threadIdx.x >> 5, lane = threadIdx.x & 31;

  const float* kbase = kv + (size_t)b * S_ * (2 * D_) + h * (2 * HD_);
  if (wv == 0) {
    tdm_load_2d((unsigned)(size_t)&tk[0], kbase, S_, HD_, 2 * D_, TDM_F32_64_);
    __builtin_amdgcn_s_wait_tensorcnt(0);
  }
  __syncthreads();

  float sw = 0.f;
#pragma unroll
  for (int q = 0; q < P_; ++q) sw += Wpos[q];
  float scale = sw * 0.125f;               // sum(Wpos) / sqrt(64)

  float* out = scores + (size_t)bh * SP_ * SP_;
  for (int t = wv; t < 7 * 14; t += 8) {
    int i0 = (half * 7 + t / 14) << 4;
    int j0 = (t % 14) << 4;

    v8f acc = {};
#pragma unroll
    for (int k0 = 0; k0 < HD_; k0 += 32) {
      v16h a  = frag_a_rowmask(tk, LDK_, i0, k0, lane, S_);
      v16h bb = frag_bt_rowmask(tk, LDK_, j0, k0, lane, S_);
      acc = __builtin_amdgcn_wmma_f32_16x16x32_f16(false, a, false, bb, (short)0, acc,
                                                   false, false);
    }

    int n  = j0 + (lane & 15);
    int mb = i0 + ((lane >> 4) << 3);
    float nmsk = (n < S_) ? 1.f : 0.f;
    int nc = imin_(n, S_ - 1);
#pragma unroll
    for (int e = 0; e < 8; ++e) {
      int m = mb + e;
      float msk = (m < S_) ? nmsk : 0.f;
      float bias = pb[imin_(m, S_ - 1) * S_ + nc] * msk;   // branchless, clamped
      out[(size_t)m * SP_ + n] = acc[e] * scale + bias;
    }
  }
}

// ---------------------------------------------------------------------------
// Row softmax over j < S; pad rows/cols zeroed. One wave per row.
// ---------------------------------------------------------------------------
__global__ void softmax_kernel(float* __restrict__ scores) {
  int row  = blockIdx.x;
  int lane = threadIdx.x;
  int i = row % SP_;
  float* p = scores + (size_t)row * SP_;

  if (i >= S_) {
    for (int j = lane; j < SP_; j += 32) p[j] = 0.f;
    return;
  }
  float mx = -1e30f;
  for (int j = lane; j < S_; j += 32) mx = fmaxf(mx, p[j]);
#pragma unroll
  for (int off = 16; off > 0; off >>= 1) mx = fmaxf(mx, __shfl_xor(mx, off, 32));

  float sum = 0.f;
  for (int j = lane; j < S_; j += 32) {
    float e = __expf(p[j] - mx);
    p[j] = e;
    sum += e;
  }
#pragma unroll
  for (int off = 16; off > 0; off >>= 1) sum += __shfl_xor(sum, off, 32);

  float inv = 1.f / sum;
  for (int j = lane; j < S_; j += 32) p[j] *= inv;
  for (int j = S_ + lane; j < SP_; j += 32) p[j] = 0.f;
}

// ---------------------------------------------------------------------------
// vals16[b,s,h*HD+d] = sum_j a[bh,s,j] * v[b,h,j,d]   (f16 output -> Wo GEMM)
// One workgroup per (b,h): TDM-stage the 200x64 f32 v panel into LDS once.
// ---------------------------------------------------------------------------
__global__ __launch_bounds__(256) void attn_vals_kernel(
    const float* __restrict__ scores, const float* __restrict__ kv,
    _Float16* __restrict__ vals16) {
  __shared__ float tv[S_ * LDK_];          // 200 x 68 f32
  int bh = blockIdx.x;                     // 0..127
  int b = bh / H_, h = bh % H_;
  int wv = threadIdx.x >> 5, lane = threadIdx.x & 31;

  const float* vbase = kv + (size_t)b * S_ * (2 * D_) + h * (2 * HD_) + HD_;
  if (wv == 0) {
    tdm_load_2d((unsigned)(size_t)&tv[0], vbase, S_, HD_, 2 * D_, TDM_F32_64_);
    __builtin_amdgcn_s_wait_tensorcnt(0);
  }
  __syncthreads();

  const float* a_ = scores + (size_t)bh * SP_ * SP_;
  const int TI = SP_ / 16;                 // 14
  const int TD = HD_ / 16;                 // 4
  for (int t = wv; t < TI * TD; t += 8) {
    int i0 = (t / TD) << 4;
    int d0 = (t % TD) << 4;

    v8f acc = {};
    for (int k0 = 0; k0 < SP_; k0 += 32) {
      v16h a  = frag_a_f32(a_, SP_, i0, k0, lane);
      v16h bb = frag_b_kmask(tv, LDK_, k0, d0, lane, S_);
      acc = __builtin_amdgcn_wmma_f32_16x16x32_f16(false, a, false, bb, (short)0, acc,
                                                   false, false);
    }

    int n  = d0 + (lane & 15);
    int mb = i0 + ((lane >> 4) << 3);
#pragma unroll
    for (int e = 0; e < 8; ++e) {
      int m = mb + e;
      if (m < S_)
        vals16[(size_t)(b * S_ + m) * D_ + h * HD_ + n] = (_Float16)acc[e];
    }
  }
}

// ---------------------------------------------------------------------------
// LayerNorm over last dim D; optional extra f16 copy of the output.
// ---------------------------------------------------------------------------
__global__ void layernorm_kernel(const float* __restrict__ X, const float* __restrict__ g,
                                 const float* __restrict__ beta, float* __restrict__ Y,
                                 _Float16* __restrict__ Y16) {
  __shared__ float sh_s[128];
  __shared__ float sh_q[128];
  int row = blockIdx.x;
  int tid = threadIdx.x;
  const float* x = X + (size_t)row * D_;
  float v[4], s = 0.f, q = 0.f;
#pragma unroll
  for (int i = 0; i < 4; ++i) {
    v[i] = x[tid + i * 128];
    s += v[i];
    q += v[i] * v[i];
  }
  sh_s[tid] = s; sh_q[tid] = q;
  __syncthreads();
  for (int off = 64; off > 0; off >>= 1) {
    if (tid < off) { sh_s[tid] += sh_s[tid + off]; sh_q[tid] += sh_q[tid + off]; }
    __syncthreads();
  }
  float mu  = sh_s[0] * (1.f / D_);
  float var = sh_q[0] * (1.f / D_) - mu * mu;
  float inv = rsqrtf(var + EPS_);
  float* y = Y + (size_t)row * D_;
#pragma unroll
  for (int i = 0; i < 4; ++i) {
    int c = tid + i * 128;
    float o = (v[i] - mu) * inv * g[c] + beta[c];
    y[c] = o;
    if (Y16) Y16[(size_t)row * D_ + c] = (_Float16)o;
  }
}

// ---------------------------------------------------------------------------
// Host launch
// ---------------------------------------------------------------------------
extern "C" void kernel_launch(void* const* d_in, const int* in_sizes, int n_in,
                              void* d_out, int out_size, void* d_ws, size_t ws_size,
                              hipStream_t stream) {
  const float* x     = (const float*)d_in[0];
  const float* pos   = (const float*)d_in[1];
  const float* Wkv   = (const float*)d_in[2];
  const float* bkv   = (const float*)d_in[3];
  const float* Wo    = (const float*)d_in[4];
  const float* bo    = (const float*)d_in[5];
  const float* Wpos  = (const float*)d_in[6];
  const float* bpos  = (const float*)d_in[7];
  const float* W1    = (const float*)d_in[8];
  const float* b1    = (const float*)d_in[9];
  const float* W2    = (const float*)d_in[10];
  const float* b2    = (const float*)d_in[11];
  const float* g1    = (const float*)d_in[12];
  const float* beta1 = (const float*)d_in[13];
  const float* g2    = (const float*)d_in[14];
  const float* beta2 = (const float*)d_in[15];
  float* out = (float*)d_out;
  float* ws  = (float*)d_ws;

  const int M = B_ * S_;                         // 3200
  // workspace layout in float-slots (f16 buffers use half a slot per element)
  const size_t off_kv     = 0;                   // f32 3,276,800
  const size_t off_pb     = 3276800;             // f32    40,000
  const size_t off_scores = 3316800;             // f32 6,422,528 -> 9,739,328
  const size_t off_vals16 = 9739328;             // f16 1,638,400 ->   819,200 slots
  const size_t off_hin    = 10558528;            // f32 1,638,400
  const size_t off_h      = 12196928;            // f32 1,638,400
  const size_t off_h16    = 13835328;            // f16 1,638,400 ->   819,200 slots
  const size_t off_ff1    = 0;                   // f16 6,553,600 -> 3,276,800 slots (aliases dead kv)
  const size_t off_ff2    = 14654528;            // f32 1,638,400
  const size_t off_x16    = 16292928;            // f16 1,638,400 ->   819,200 slots
  const size_t off_wkv16  = 17112128;            // f16 524,288   ->   262,144 slots
  const size_t off_wo16   = 17374272;            // f16 262,144   ->   131,072 slots
  const size_t off_w116   = 17505344;            // f16 1,048,576 ->   524,288 slots
  const size_t off_w216   = 18029632;            // f16 1,048,576 ->   524,288 slots -> 18,553,920 total

  float* kv     = ws + off_kv;
  float* pb     = ws + off_pb;
  float* scores = ws + off_scores;
  float* hin    = ws + off_hin;
  float* h      = ws + off_h;
  float* ff2in  = ws + off_ff2;
  _Float16* vals16 = (_Float16*)(ws + off_vals16);
  _Float16* h16    = (_Float16*)(ws + off_h16);
  _Float16* ff1_16 = (_Float16*)(ws + off_ff1);
  _Float16* x16    = (_Float16*)(ws + off_x16);
  _Float16* wkv16  = (_Float16*)(ws + off_wkv16);
  _Float16* wo16   = (_Float16*)(ws + off_wo16);
  _Float16* w116   = (_Float16*)(ws + off_w116);
  _Float16* w216   = (_Float16*)(ws + off_w216);

  // 0) pack f32 operands to f16 (one-time per call)
  cvt_f16_kernel<<<(M * D_ / 4 + 255) / 256, 256, 0, stream>>>(x, x16, M * D_);
  cvt_f16_kernel<<<(2 * D_ * D_ / 4 + 255) / 256, 256, 0, stream>>>(Wkv, wkv16, 2 * D_ * D_);
  cvt_f16_kernel<<<(D_ * D_ / 4 + 255) / 256, 256, 0, stream>>>(Wo, wo16, D_ * D_);
  cvt_f16_kernel<<<(FF_ * D_ / 4 + 255) / 256, 256, 0, stream>>>(W1, w116, FF_ * D_);
  cvt_f16_kernel<<<(D_ * FF_ / 4 + 255) / 256, 256, 0, stream>>>(W2, w216, D_ * FF_);

  // 1) positional bias
  pos_bias_kernel<<<(S_ * S_ + 255) / 256, 256, 0, stream>>>(pos, Wpos, bpos, pb);

  // 2) kv = x @ Wkv^T + bkv            (3200 x 1024, K=512)
  wmma_gemm_nt<false, false><<<(M / BM_) * (2 * D_ / BN_), 256, 0, stream>>>(
      x16, wkv16, bkv, nullptr, kv, M, 2 * D_, D_);

  // 3) attention scores (+ scale + pos bias), padded SP x SP
  attn_scores_kernel<<<B_ * H_ * 2, 256, 0, stream>>>(kv, pb, Wpos, scores);

  // 4) softmax per row
  softmax_kernel<<<B_ * H_ * SP_, 32, 0, stream>>>(scores);

  // 5) vals16 = a @ v (f16)
  attn_vals_kernel<<<B_ * H_, 256, 0, stream>>>(scores, kv, vals16);

  // 6) hin = vals @ Wo^T + bo + x      (3200 x 512, K=512)
  wmma_gemm_nt<false, false><<<(M / BM_) * (D_ / BN_), 256, 0, stream>>>(
      vals16, wo16, bo, x, hin, M, D_, D_);

  // 7) h = LayerNorm(hin)  (+ f16 copy for the FFN GEMM)
  layernorm_kernel<<<M, 128, 0, stream>>>(hin, g1, beta1, h, h16);

  // 8) ff1 = relu(h @ W1^T + b1)       (3200 x 2048, K=512), f16 output
  wmma_gemm_nt<true, true><<<(M / BM_) * (FF_ / BN_), 256, 0, stream>>>(
      h16, w116, b1, nullptr, ff1_16, M, FF_, D_);

  // 9) ff2in = ff1 @ W2^T + b2 + h     (3200 x 512, K=2048)
  wmma_gemm_nt<false, false><<<(M / BM_) * (D_ / BN_), 256, 0, stream>>>(
      ff1_16, w216, b2, h, ff2in, M, D_, FF_);

  // 10) out = LayerNorm(ff2in)
  layernorm_kernel<<<M, 128, 0, stream>>>(ff2in, g2, beta2, out, nullptr);
}
